// SSMBlock_14190571946380
// MI455X (gfx1250) — compile-verified, matrix operand order checked by reading, not run on hardware
//
#include <hip/hip_runtime.h>
#include <hip/hip_bf16.h>
#include <cstdint>

#define D_MODEL 1024
#define D_STATE 16
#define D_INNER 2048
#define KSIZE   4
#define BB      2
#define LL      2048
#define NROWS   (BB * LL)            // 4096 token rows
#define NXZ     (2 * D_INNER)        // 4096

typedef __attribute__((ext_vector_type(16))) __bf16 v16bf;
typedef __attribute__((ext_vector_type(8)))  __bf16 v8bf;
typedef __attribute__((ext_vector_type(8)))  float  v8f;

// ---------------------------------------------------------------------------
// CDNA5 async global->LDS copy (ASYNCcnt path), with sync fallback.
// Builtin prototype (from hipcc diagnostic): param0 = AS(1) int4* (non-const).
// ---------------------------------------------------------------------------
#if defined(__has_builtin)
#  if __has_builtin(__builtin_amdgcn_global_load_async_to_lds_b128)
#    define HAVE_ASYNC_COPY 1
#  endif
#  if __has_builtin(__builtin_amdgcn_s_wait_asynccnt)
#    define HAVE_WAIT_ASYNC 1
#  endif
#endif
#ifndef HAVE_ASYNC_COPY
#  define HAVE_ASYNC_COPY 0
#endif
#ifndef HAVE_WAIT_ASYNC
#  define HAVE_WAIT_ASYNC 0
#endif

typedef int v4i_ __attribute__((vector_size(16)));
typedef __attribute__((address_space(1))) v4i_ as1_v4i;
typedef __attribute__((address_space(3))) v4i_ as3_v4i;

__device__ __forceinline__ void async_copy16(const void* g, void* l) {
#if HAVE_ASYNC_COPY
  // Generic->AS(1): flat global addr == global addr. Generic->AS(3): low 32
  // bits of a flat LDS address are the LDS byte offset (ISA 10.2).
  __builtin_amdgcn_global_load_async_to_lds_b128(
      (as1_v4i*)(uintptr_t)g, (as3_v4i*)(uintptr_t)l, 0, 0);
#else
  *(uint4*)l = *(const uint4*)g;
#endif
}

#if HAVE_ASYNC_COPY
#  if HAVE_WAIT_ASYNC
#    define WAIT_ASYNC(n) __builtin_amdgcn_s_wait_asynccnt(n)
#  else
#    define WAIT_ASYNC(n) asm volatile("s_wait_asynccnt %0" ::"n"(n) : "memory")
#  endif
#else
#  define WAIT_ASYNC(n) ((void)0)
#endif

// ---------------------------------------------------------------------------
// WMMA 16-bit operand fragment: lane l (m/n = l&15, khalf = l>>4) holds
// k = khalf*8 + [0..7] in elements 0..7 and k = 16+khalf*8+[0..7] in 8..15.
// Two b128 LDS reads; 80B row stride -> banks 20*m mod 64, conflict-free.
// ---------------------------------------------------------------------------
__device__ __forceinline__ v16bf make_frag(const __bf16* rowbase, int khalf) {
  v8bf lo = *(const v8bf*)(rowbase + khalf * 8);
  v8bf hi = *(const v8bf*)(rowbase + 16 + khalf * 8);
  v16bf r;
#pragma unroll
  for (int i = 0; i < 8; ++i) { r[i] = lo[i]; r[i + 8] = hi[i]; }
  return r;
}

// ---------------------------------------------------------------------------
// C(MxN,f32) = A(MxK,bf16,row-major) * Bt(NxK,bf16,row-major)^T
// Block tile 64(M) x 128(N), BK=32, double-buffered async LDS staging.
// 8 waves as 2(M) x 4(N); each wave: 2x2 16x16 tiles = 4 WMMA per stage.
// ---------------------------------------------------------------------------
__global__ __launch_bounds__(256) void gemm_bf16_wmma(
    const __bf16* __restrict__ A, const __bf16* __restrict__ Bt,
    float* __restrict__ C, int M, int N, int K) {
  __shared__ __bf16 As[2][64][40];
  __shared__ __bf16 Bs[2][128][40];

  const int t     = threadIdx.x;
  const int lane  = t & 31;
  const int w     = t >> 5;
  const int m0    = blockIdx.y * 64;
  const int n0    = blockIdx.x * 128;
  const int wm    = (w >> 2) * 32;   // 0 / 32
  const int wn    = (w & 3) * 32;    // 0 / 32 / 64 / 96
  const int mrow  = lane & 15;
  const int khalf = lane >> 4;

  const v8f z8 = {0.f, 0.f, 0.f, 0.f, 0.f, 0.f, 0.f, 0.f};
  v8f acc[2][2] = {{z8, z8}, {z8, z8}};

  // staging: each thread copies one 16B chunk of A and two of B per stage
  const int srow = t >> 2;          // 0..63
  const int schk = (t & 3) * 8;     // bf16 element offset 0/8/16/24

  auto stage = [&](int buf, int kb) {
    async_copy16(&A[(size_t)(m0 + srow) * K + kb + schk], &As[buf][srow][schk]);
    async_copy16(&Bt[(size_t)(n0 + srow) * K + kb + schk], &Bs[buf][srow][schk]);
    async_copy16(&Bt[(size_t)(n0 + 64 + srow) * K + kb + schk],
                 &Bs[buf][64 + srow][schk]);
  };

  const int nk = K / 32;
  stage(0, 0);
  for (int i = 0; i < nk; ++i) {
    const int cur = i & 1;
    if (i + 1 < nk) {
      stage(cur ^ 1, (i + 1) * 32);
      WAIT_ASYNC(3);                 // in-order: stage i's 3 copies retired
    } else {
      WAIT_ASYNC(0);
    }
    __syncthreads();                 // all waves' stage-i copies visible

    v16bf a0 = make_frag(&As[cur][wm + mrow][0], khalf);
    v16bf a1 = make_frag(&As[cur][wm + 16 + mrow][0], khalf);
    v16bf b0 = make_frag(&Bs[cur][wn + mrow][0], khalf);
    v16bf b1 = make_frag(&Bs[cur][wn + 16 + mrow][0], khalf);
    acc[0][0] = __builtin_amdgcn_wmma_f32_16x16x32_bf16(false, a0, false, b0,
                                                        (short)0, acc[0][0], false, false);
    acc[0][1] = __builtin_amdgcn_wmma_f32_16x16x32_bf16(false, a0, false, b1,
                                                        (short)0, acc[0][1], false, false);
    acc[1][0] = __builtin_amdgcn_wmma_f32_16x16x32_bf16(false, a1, false, b0,
                                                        (short)0, acc[1][0], false, false);
    acc[1][1] = __builtin_amdgcn_wmma_f32_16x16x32_bf16(false, a1, false, b1,
                                                        (short)0, acc[1][1], false, false);
    __syncthreads();                 // buf 'cur' free for stage i+2
  }

  // C/D layout: VGPR e -> M = e + 8*(lane>>4); N = lane&15.
  const int grp = lane >> 4;
#pragma unroll
  for (int mi = 0; mi < 2; ++mi)
#pragma unroll
    for (int ni = 0; ni < 2; ++ni)
#pragma unroll
      for (int e = 0; e < 8; ++e)
        C[(size_t)(m0 + wm + mi * 16 + grp * 8 + e) * N +
          (n0 + wn + ni * 16 + mrow)] = acc[mi][ni][e];
}

// ---------------------------------------------------------------------------
// f32 -> bf16 elementwise (vectorized 4-wide)
// ---------------------------------------------------------------------------
__global__ __launch_bounds__(256) void cvt_bf16_kernel(
    const float* __restrict__ in, __bf16* __restrict__ out, int n4) {
  const int idx = blockIdx.x * 256 + threadIdx.x;
  if (idx >= n4) return;
  const float4 v = ((const float4*)in)[idx];
  union { __bf16 b[4]; uint2 u; } p;
  p.b[0] = (__bf16)v.x; p.b[1] = (__bf16)v.y;
  p.b[2] = (__bf16)v.z; p.b[3] = (__bf16)v.w;
  ((uint2*)out)[idx] = p.u;
}

// ---------------------------------------------------------------------------
// Wt(NxK,bf16) = transpose(W(KxN,f32)); 32x32 LDS tile, one-shot weight prep.
// ---------------------------------------------------------------------------
__global__ __launch_bounds__(256) void transpose_cvt_kernel(
    const float* __restrict__ W, __bf16* __restrict__ Wt, int K, int N) {
  __shared__ float tile[32][33];
  const int k0 = blockIdx.y * 32, n0 = blockIdx.x * 32;
  const int r = threadIdx.x >> 3, c4 = (threadIdx.x & 7) * 4;
  const float4 v = *(const float4*)&W[(size_t)(k0 + r) * N + n0 + c4];
  tile[r][c4 + 0] = v.x; tile[r][c4 + 1] = v.y;
  tile[r][c4 + 2] = v.z; tile[r][c4 + 3] = v.w;
  __syncthreads();
  union { __bf16 b[4]; uint2 u; } p;
#pragma unroll
  for (int i = 0; i < 4; ++i) p.b[i] = (__bf16)tile[c4 + i][r];
  *(uint2*)&Wt[(size_t)(n0 + r) * K + k0 + c4] = p.u;
}

__device__ __forceinline__ float silu_f(float v) { return v / (1.f + __expf(-v)); }
__device__ __forceinline__ float softplus_f(float v) {
  return (v > 20.f) ? v : log1pf(__expf(v));
}

// ---------------------------------------------------------------------------
// Causal depthwise conv (K=4) + bias + SiLU; x half read from xz (f32).
// ---------------------------------------------------------------------------
__global__ __launch_bounds__(256) void conv_silu_kernel(
    const float* __restrict__ xz, const float* __restrict__ Wc,
    const float* __restrict__ bc, float* __restrict__ x_in) {
  const int idx = blockIdx.x * 256 + threadIdx.x;
  if (idx >= NROWS * D_INNER) return;
  const int c   = idx & (D_INNER - 1);
  const int row = idx >> 11;                 // b*L + l
  const int l   = row & (LL - 1);
  float acc = bc[c];
#pragma unroll
  for (int k = 0; k < KSIZE; ++k) {
    const int ll = l - (KSIZE - 1) + k;
    if (ll >= 0)
      acc += Wc[c * KSIZE + k] * xz[(size_t)(row - (KSIZE - 1) + k) * NXZ + c];
  }
  x_in[(size_t)row * D_INNER + c] = silu_f(acc);
}

// ---------------------------------------------------------------------------
// xBC = x_in @ W_x (K=2048, N=33) -> dt_raw / B_ssm / C_ssm.
// ---------------------------------------------------------------------------
__global__ __launch_bounds__(256) void xbc_kernel(
    const float* __restrict__ x_in, const float* __restrict__ W_x,
    float* __restrict__ dtraw, float* __restrict__ Bs, float* __restrict__ Cs) {
  __shared__ float red[33][257];
  const int row = blockIdx.x;
  const int t   = threadIdx.x;
  float part[33];
#pragma unroll
  for (int j = 0; j < 33; ++j) part[j] = 0.f;
  for (int c = t; c < D_INNER; c += 256) {
    const float xv = x_in[(size_t)row * D_INNER + c];
#pragma unroll
    for (int j = 0; j < 33; ++j) part[j] += xv * W_x[c * 33 + j];
  }
#pragma unroll
  for (int j = 0; j < 33; ++j) red[j][t] = part[j];
  __syncthreads();
  for (int s = 128; s > 0; s >>= 1) {
    if (t < s)
#pragma unroll
      for (int j = 0; j < 33; ++j) red[j][t] += red[j][t + s];
    __syncthreads();
  }
  if (t < 33) {
    const float v = red[t][0];
    if (t == 0)      dtraw[row] = v;
    else if (t < 17) Bs[row * D_STATE + t - 1]  = v;
    else             Cs[row * D_STATE + t - 17] = v;
  }
}

// ---------------------------------------------------------------------------
// Selective scan fused with dt=softplus, +D*x skip, SiLU(z) gate; one thread
// per (d,n) recurrence (65536 lanes -> 256 blocks); 64-step LDS chunks;
// y = sum_n h*C via 16-lane shfl_xor groups. Emits s directly in bf16.
// ---------------------------------------------------------------------------
__global__ __launch_bounds__(256) void scan_kernel(
    const float* __restrict__ xz, const float* __restrict__ x_in,
    const float* __restrict__ dtraw, const float* __restrict__ Bs,
    const float* __restrict__ Cs, const float* __restrict__ A_log,
    const float* __restrict__ W_dt, const float* __restrict__ b_dt,
    const float* __restrict__ Dvec, __bf16* __restrict__ s_bf) {
  __shared__ float xs[64][16], zs[64][16], Bsl[64][16], Csl[64][16], ys[64][16];
  __shared__ float dts[64];

  const int t  = threadIdx.x;
  const int n  = t & 15;
  const int dl = t >> 4;
  const int b  = blockIdx.x >> 7;
  const int d0 = (blockIdx.x & 127) * 16;
  const int d  = d0 + dl;

  const float Ac   = -__expf(A_log[d * D_STATE + n]);
  const float Wdtd = W_dt[d];
  const float bdtd = b_dt[d];
  const float Dd   = Dvec[d];

  float h = 0.f;
  for (int lc = 0; lc < LL; lc += 64) {
    for (int idx = t; idx < 64 * 16; idx += 256) {
      const int i = idx >> 4, j = idx & 15;
      const size_t row = (size_t)(b * LL + lc + i);
      xs[i][j]  = x_in[row * D_INNER + d0 + j];
      zs[i][j]  = xz[row * NXZ + D_INNER + d0 + j];
      Bsl[i][j] = Bs[row * D_STATE + j];
      Csl[i][j] = Cs[row * D_STATE + j];
    }
    if (t < 64) dts[t] = dtraw[b * LL + lc + t];
    __syncthreads();

    for (int i = 0; i < 64; ++i) {
      const float dt_t = softplus_f(dts[i] * Wdtd + bdtd);
      const float xv   = xs[i][dl];
      const float dA   = __expf(dt_t * Ac);
      h = dA * h + (dt_t * Bsl[i][n]) * xv;
      float p = h * Csl[i][n];
#pragma unroll
      for (int m = 8; m > 0; m >>= 1) p += __shfl_xor(p, m, 16);
      if (n == 0) {
        const float zv = zs[i][dl];
        ys[i][dl] = (p + Dd * xv) * silu_f(zv);
      }
    }
    __syncthreads();

    for (int idx = t; idx < 64 * 16; idx += 256) {
      const int i = idx >> 4, j = idx & 15;
      s_bf[(size_t)(b * LL + lc + i) * D_INNER + d0 + j] = (__bf16)ys[i][j];
    }
    __syncthreads();
  }
}

// ---------------------------------------------------------------------------
extern "C" void kernel_launch(void* const* d_in, const int* in_sizes, int n_in,
                              void* d_out, int out_size, void* d_ws, size_t ws_size,
                              hipStream_t stream) {
  const float* x      = (const float*)d_in[0];
  const float* W_in   = (const float*)d_in[1];
  const float* W_conv = (const float*)d_in[2];
  const float* b_conv = (const float*)d_in[3];
  const float* W_x    = (const float*)d_in[4];
  const float* W_dt   = (const float*)d_in[5];
  const float* b_dt   = (const float*)d_in[6];
  const float* W_out  = (const float*)d_in[7];
  const float* A_log  = (const float*)d_in[8];
  const float* Dvec   = (const float*)d_in[9];
  float* out = (float*)d_out;

  float* ws    = (float*)d_ws;
  float* xz    = ws;                                    // 4096*4096 f32
  float* x_in  = xz + (size_t)NROWS * NXZ;              // 4096*2048 f32
  float* dtraw = x_in + (size_t)NROWS * D_INNER;        // 4096
  float* Bssm  = dtraw + NROWS;                         // 4096*16
  float* Cssm  = Bssm + NROWS * D_STATE;                // 4096*16
  __bf16* x_bf   = (__bf16*)(Cssm + NROWS * D_STATE);   // 4096*1024
  __bf16* win_t  = x_bf + (size_t)NROWS * D_MODEL;      // 4096*1024
  __bf16* wout_t = win_t + (size_t)NXZ * D_MODEL;       // 1024*2048
  __bf16* s_bf   = wout_t + (size_t)D_MODEL * D_INNER;  // 4096*2048

  // 0) operand prep: x -> bf16; W_in, W_out -> transposed bf16
  cvt_bf16_kernel<<<(NROWS * D_MODEL / 4 + 255) / 256, 256, 0, stream>>>(
      x, x_bf, NROWS * D_MODEL / 4);
  transpose_cvt_kernel<<<dim3(NXZ / 32, D_MODEL / 32), 256, 0, stream>>>(
      W_in, win_t, D_MODEL, NXZ);
  transpose_cvt_kernel<<<dim3(D_MODEL / 32, D_INNER / 32), 256, 0, stream>>>(
      W_out, wout_t, D_INNER, D_MODEL);

  // 1) xz = x @ W_in   (M=4096, N=4096, K=1024)
  gemm_bf16_wmma<<<dim3(NXZ / 128, NROWS / 64), 256, 0, stream>>>(
      x_bf, win_t, xz, NROWS, NXZ, D_MODEL);

  // 2) causal depthwise conv + SiLU
  conv_silu_kernel<<<(NROWS * D_INNER) / 256, 256, 0, stream>>>(
      xz, W_conv, b_conv, x_in);

  // 3) xBC projection -> dt_raw / B_ssm / C_ssm
  xbc_kernel<<<NROWS, 256, 0, stream>>>(x_in, W_x, dtraw, Bssm, Cssm);

  // 4) fused selective scan + skip + gate -> s (bf16)
  scan_kernel<<<BB * (D_INNER / 16), 256, 0, stream>>>(
      xz, x_in, dtraw, Bssm, Cssm, A_log, W_dt, b_dt, Dvec, s_bf);

  // 5) out = s @ W_out  (M=4096, N=1024, K=2048)
  gemm_bf16_wmma<<<dim3(D_MODEL / 128, NROWS / 64), 256, 0, stream>>>(
      s_bf, wout_t, out, NROWS, D_MODEL, D_INNER);
}
